// FusionSeparateDecoder_8899172237588
// MI455X (gfx1250) — matrix operand order repaired and checked by reading, not compiled, not run on hardware
//
#include <hip/hip_runtime.h>
#include <hip/hip_bf16.h>

typedef __attribute__((ext_vector_type(16))) _Float16 v16h;
typedef __attribute__((ext_vector_type(8)))  float    v8f;
typedef __attribute__((ext_vector_type(8)))  unsigned v8u;

#define NG    128        // graphs
#define NPG   256        // nodes per graph
#define CIN   64
#define HID   64
#define NROWS (NG * NPG) // 32768

// Workspace layout (f16 elements):
//   [0,            6*4096)              : Wt (transposed f16 weights) -- W1t mlp0..2 then W2t mlp0..2
//   [6*4096,       6*4096 + 3*NROWS*64) : F16 features fa,fb,ft
#define WS_W_F16   (6 * 4096)
#define WS_F_BASE  WS_W_F16

__device__ __forceinline__ unsigned pack2h(float x, float y) {
    _Float16 a = (_Float16)x, b = (_Float16)y;
    unsigned short ua = __builtin_bit_cast(unsigned short, a);
    unsigned short ub = __builtin_bit_cast(unsigned short, b);
    return ((unsigned)ub << 16) | (unsigned)ua;
}
__device__ __forceinline__ unsigned short h16(float x) {
    return __builtin_bit_cast(unsigned short, (_Float16)x);
}

// ---------------------------------------------------------------------------
// Kernel 1: transpose+convert the six 64x64 weight matrices to f16 Wt[n][k].
// (z @ W means sum_k z[k]*W[k][n]; B-fragment wants K contiguous per column n.)
// ---------------------------------------------------------------------------
__global__ void prep_weights_kernel(const float* Wa1, const float* Wb1, const float* Wt1,
                                    const float* Wa2, const float* Wb2, const float* Wt2,
                                    unsigned short* wsW) {
    int idx = blockIdx.x * blockDim.x + threadIdx.x;
    if (idx >= 6 * 4096) return;
    int m = idx / 4096;        // 0..2 = layer-1 of mlp m, 3..5 = layer-2 of mlp m-3
    int e = idx % 4096;
    int n = e / 64, k = e % 64;
    const float* W;
    switch (m) {
        case 0: W = Wa1; break; case 1: W = Wb1; break; case 2: W = Wt1; break;
        case 3: W = Wa2; break; case 4: W = Wb2; break; default: W = Wt2; break;
    }
    wsW[idx] = h16(W[k * 64 + n]);   // Wt[n][k] = W[k][n]
}

// ---------------------------------------------------------------------------
// Kernel 2: 3 MLPs via WMMA. 4 waves/block, 1 wave = one 16-row tile of X.
// ---------------------------------------------------------------------------
#define LDS_STRIDE 66   // ushorts per 64-col row (even -> 32-bit aligned pairs)

__global__ __launch_bounds__(128) void mlp_kernel(
    const float* __restrict__ z,
    const float* __restrict__ ba1, const float* __restrict__ ba2,
    const float* __restrict__ bb1, const float* __restrict__ bb2,
    const float* __restrict__ bt1, const float* __restrict__ bt2,
    const unsigned short* __restrict__ wsW,
    unsigned short* __restrict__ wsF,
    float* __restrict__ fOut) {

    __shared__ __align__(16) unsigned short sH[4 * 16 * LDS_STRIDE];

    const int tid  = threadIdx.x;
    const int lane = tid & 31;
    const int wave = tid >> 5;
    const int l15  = lane & 15;
    const int hi   = lane >> 4;          // 0 = lanes 0-15, 1 = lanes 16-31
    const int tile = blockIdx.x * 4 + wave;
    const int r0   = tile * 16;

    unsigned short* sW16 = &sH[wave * 16 * LDS_STRIDE];
    unsigned*       sWu  = (unsigned*)sW16;         // 33 uints per row

    // --- Build A fragments of z once (shared by all 3 MLPs). f32 -> f16.
    // A 16-bit layout: lane holds row M=l15; VGPR p: K = (p<4?0:16) + hi*8 + 2*(p&3).
    v16h Az[2];
    {
        const float* zrow = z + (size_t)(r0 + l15) * CIN;
        #pragma unroll
        for (int kc = 0; kc < 2; ++kc) {
            v8u u;
            #pragma unroll
            for (int p = 0; p < 8; ++p) {
                int k = kc * 32 + ((p < 4) ? 0 : 16) + hi * 8 + 2 * (p & 3);
                float2 v = *(const float2*)(zrow + k);
                u[p] = pack2h(v.x, v.y);
            }
            Az[kc] = __builtin_bit_cast(v16h, u);
        }
    }

    for (int m = 0; m < 3; ++m) {
        const unsigned* W1u = (const unsigned*)wsW + (size_t)m * 2048;        // 4096 f16
        const unsigned* W2u = (const unsigned*)wsW + (size_t)(3 + m) * 2048;
        const float* b1 = (m == 0) ? ba1 : ((m == 1) ? bb1 : bt1);
        const float* b2 = (m == 0) ? ba2 : ((m == 1) ? bb2 : bt2);

        // ---- Layer 1: h = relu(z @ W1 + b1); write h (f16) to LDS tile.
        #pragma unroll
        for (int nt = 0; nt < 4; ++nt) {
            v8f c = {};
            #pragma unroll
            for (int kc = 0; kc < 2; ++kc) {
                // B layout: lane holds col N=l15; VGPR p: K = (p<4?0:16) + hi*8 + 2*(p&3)  (+ kc*32)
                v8u u;
                #pragma unroll
                for (int p = 0; p < 8; ++p) {
                    int kp = kc * 16 + ((p < 4) ? 0 : 8) + hi * 4 + (p & 3);           // uint (K-pair) index
                    u[p] = W1u[(nt * 16 + l15) * 32 + kp];
                }
                v16h bf = __builtin_bit_cast(v16h, u);
                c = __builtin_amdgcn_wmma_f32_16x16x32_f16(
                        false, Az[kc], false, bf, (short)0, c, false, false);
            }
            float bv = b1[nt * 16 + l15];
            #pragma unroll
            for (int j = 0; j < 8; ++j) {                    // C layout: row = hi*8+j
                float h = c[j] + bv; h = h > 0.f ? h : 0.f;
                sW16[(hi * 8 + j) * LDS_STRIDE + nt * 16 + l15] = h16(h);
            }
        }
        __syncthreads();

        // ---- Re-layout h as A fragments for layer 2.
        v16h Ah[2];
        #pragma unroll
        for (int kc = 0; kc < 2; ++kc) {
            v8u u;
            #pragma unroll
            for (int p = 0; p < 8; ++p) {
                int kp = kc * 16 + ((p < 4) ? 0 : 8) + hi * 4 + (p & 3);
                u[p] = sWu[l15 * 33 + kp];
            }
            Ah[kc] = __builtin_bit_cast(v16h, u);
        }
        __syncthreads();

        // ---- Layer 2: f = relu(h @ W2 + b2); f32 to d_out (NT), f16 to LDS -> ws.
        float* fo = fOut + (size_t)m * NROWS * HID;
        #pragma unroll
        for (int nt = 0; nt < 4; ++nt) {
            v8f c = {};
            #pragma unroll
            for (int kc = 0; kc < 2; ++kc) {
                v8u u;
                #pragma unroll
                for (int p = 0; p < 8; ++p) {
                    int kp = kc * 16 + ((p < 4) ? 0 : 8) + hi * 4 + (p & 3);
                    u[p] = W2u[(nt * 16 + l15) * 32 + kp];
                }
                v16h bf = __builtin_bit_cast(v16h, u);
                c = __builtin_amdgcn_wmma_f32_16x16x32_f16(
                        false, Ah[kc], false, bf, (short)0, c, false, false);
            }
            float bv = b2[nt * 16 + l15];
            #pragma unroll
            for (int j = 0; j < 8; ++j) {
                float f = c[j] + bv; f = f > 0.f ? f : 0.f;
                int row = hi * 8 + j, col = nt * 16 + l15;
                // streaming f32 output: never re-read -> non-temporal store
                __builtin_nontemporal_store(f, &fo[(size_t)(r0 + row) * HID + col]);
                sW16[row * LDS_STRIDE + col] = h16(f);
            }
        }
        __syncthreads();

        // ---- Coalesced f16 copy of the 16x64 tile to workspace (re-read by
        //      gram kernel; 12.6 MB total -> keep regular-temporal, lives in L2).
        unsigned* Fu = (unsigned*)wsF + ((size_t)m * NROWS + r0) * 32;
        #pragma unroll
        for (int idx = 0; idx < 16; ++idx) {
            int e = idx * 32 + lane;                // 512 uints / 32 lanes
            Fu[e] = sWu[(e >> 5) * 33 + (e & 31)];
        }
        __syncthreads();
    }
}

// ---------------------------------------------------------------------------
// Kernel 3: adj = sigmoid(F F^T) per (graph, feature). One block per (g,t),
// F (256x64 f16 = 32KB) staged in LDS, 8 waves x 32 tiles of 16x16.
// ---------------------------------------------------------------------------
__global__ __launch_bounds__(256) void gram_kernel(
    const unsigned short* __restrict__ wsF, float* __restrict__ out) {

    __shared__ __align__(16) unsigned short sF[NPG * LDS_STRIDE];   // ~33.8 KB

    const int b = blockIdx.x;
    const int g = b & (NG - 1);
    const int t = b >> 7;

    // Stage F tile (coalesced 32-bit loads; should mostly hit L2).
    const unsigned* Fu  = (const unsigned*)wsF + ((size_t)t * NROWS + (size_t)g * NPG) * 32;
    unsigned*       sFu = (unsigned*)sF;
    for (int idx = threadIdx.x; idx < NPG * 32; idx += 256)
        sFu[(idx >> 5) * 33 + (idx & 31)] = Fu[idx];
    __syncthreads();

    const int lane = threadIdx.x & 31;
    const int wave = threadIdx.x >> 5;
    const int l15  = lane & 15;
    const int hi   = lane >> 4;

    float* adj = out + (size_t)t * NG * NPG * NPG + (size_t)g * NPG * NPG;

    for (int s = 0; s < 32; ++s) {
        int tileIdx = s * 8 + wave;
        int ti = tileIdx >> 4;                 // output row tile
        int tj = tileIdx & 15;                 // output col tile
        v8f c = {};
        #pragma unroll
        for (int kc = 0; kc < 2; ++kc) {
            v8u ua, ub;
            #pragma unroll
            for (int p = 0; p < 8; ++p) {
                // A: row = ti*16 + l15; K pair per A layout (b128-mergeable halves)
                int kpa = kc * 16 + ((p < 4) ? 0 : 8) + hi * 4 + (p & 3);
                ua[p] = sFu[(ti * 16 + l15) * 33 + kpa];
                // B = F^T: element (k, n) = F[tj*16+n][k]; K pair per B layout
                int kpb = kc * 16 + ((p < 4) ? 0 : 8) + hi * 4 + (p & 3);
                ub[p] = sFu[(tj * 16 + l15) * 33 + kpb];
            }
            v16h av = __builtin_bit_cast(v16h, ua);
            v16h bv = __builtin_bit_cast(v16h, ub);
            c = __builtin_amdgcn_wmma_f32_16x16x32_f16(
                    false, av, false, bv, (short)0, c, false, false);
        }
        #pragma unroll
        for (int j = 0; j < 8; ++j) {
            float x   = c[j];
            float sig = 1.0f / (1.0f + __expf(-x));
            int row = ti * 16 + hi * 8 + j;
            int col = tj * 16 + l15;
            // 100 MB write-only stream -> non-temporal, don't pollute L2
            __builtin_nontemporal_store(sig, &adj[(size_t)row * NPG + col]);
        }
    }
}

// ---------------------------------------------------------------------------
extern "C" void kernel_launch(void* const* d_in, const int* in_sizes, int n_in,
                              void* d_out, int out_size, void* d_ws, size_t ws_size,
                              hipStream_t stream) {
    const float* z   = (const float*)d_in[0];
    // d_in[1] = batch (uniform segments -> unused)
    const float* Wa1 = (const float*)d_in[2];  const float* ba1 = (const float*)d_in[3];
    const float* Wa2 = (const float*)d_in[4];  const float* ba2 = (const float*)d_in[5];
    const float* Wb1 = (const float*)d_in[6];  const float* bb1 = (const float*)d_in[7];
    const float* Wb2 = (const float*)d_in[8];  const float* bb2 = (const float*)d_in[9];
    const float* Wt1 = (const float*)d_in[10]; const float* bt1 = (const float*)d_in[11];
    const float* Wt2 = (const float*)d_in[12]; const float* bt2 = (const float*)d_in[13];

    unsigned short* wsW = (unsigned short*)d_ws;
    unsigned short* wsF = wsW + WS_F_BASE;

    float* out  = (float*)d_out;                       // adj_a | adj_b | adj_t | fa | fb | ft
    float* fOut = out + (size_t)3 * NG * NPG * NPG;

    prep_weights_kernel<<<96, 256, 0, stream>>>(Wa1, Wb1, Wt1, Wa2, Wb2, Wt2, wsW);
    mlp_kernel<<<NROWS / 16 / 4, 128, 0, stream>>>(z, ba1, ba2, bb1, bb2, bt1, bt2,
                                                   wsW, wsF, fOut);
    gram_kernel<<<NG * 3, 256, 0, stream>>>(wsF, out);
}
